// GRNLayer_24979529793773
// MI455X (gfx1250) — compile-verified, matrix operand order checked by reading, not run on hardware
//
#include <hip/hip_runtime.h>
#include <math.h>

// ---------------------------------------------------------------- constants
#define BB   64
#define NN   128
#define DD   512
#define TD   1536           // 3*D
#define BN   (BB*NN)        // 8192 rows for precompute GEMMs
#define NEGINF (-1.0e30f)

typedef __attribute__((ext_vector_type(16))) _Float16 v16h;
typedef __attribute__((ext_vector_type(8)))  _Float16 v8h;
typedef __attribute__((ext_vector_type(8)))  float    v8f;
typedef __attribute__((ext_vector_type(4)))  int      v4i;

#define AS1 __attribute__((address_space(1)))
#define AS3 __attribute__((address_space(3)))

#if __has_builtin(__builtin_amdgcn_global_load_async_to_lds_b128) && \
    __has_builtin(__builtin_amdgcn_s_wait_asynccnt)
#define USE_ASYNC_LDS 1
#else
#define USE_ASYNC_LDS 0
#endif

__device__ __forceinline__ AS1 v4i* as_g4(const void* p) {
    return (AS1 v4i*)(uintptr_t)p;                     // global generic == as(1) numerically
}
__device__ __forceinline__ AS3 v4i* as_l4(const void* p) {
    return (AS3 v4i*)(uint32_t)(uintptr_t)p;           // low 32 bits of generic LDS addr = LDS offset
}

__device__ __forceinline__ float sigm(float x) { return 1.0f / (1.0f + __expf(-x)); }

// WMMA: D = A(16x32 f16) * B(32x16 f16) + C(16x16 f32)
__device__ __forceinline__ v8f wmma32(v16h a, v16h b, v8f c) {
    return __builtin_amdgcn_wmma_f32_16x16x32_f16(false, a, false, b, (short)0, c, false, false);
}

// A-fragment from a row-major f16 matrix (global or LDS), ISA 16-bit A layout:
// lane<16: V0..3 hold K = k0+0..7,  V4..7 hold K = k0+16..23
// lane>=16: V0..3 hold K = k0+8..15, V4..7 hold K = k0+24..31 ; row = lane&15
__device__ __forceinline__ v16h afrag(const _Float16* A, int lda, int row0, int k0, int lane) {
    const _Float16* p = A + (size_t)(row0 + (lane & 15)) * lda + k0 + ((lane & 16) ? 8 : 0);
    v8h lo = *(const v8h*)p;
    v8h hi = *(const v8h*)(p + 16);
    v16h r;
#pragma unroll
    for (int j = 0; j < 8; ++j) { r[j] = lo[j]; r[j + 8] = hi[j]; }
    return r;
}

// B-fragment from the pre-swizzled weight buffer: one contiguous 32B per lane.
__device__ __forceinline__ v16h bfrag(const _Float16* Bsw, int NT, int kt, int ct, int lane) {
    return *(const v16h*)(Bsw + (((size_t)(kt * NT + ct)) * 32 + lane) * 16);
}

// ---------------------------------------------------------------- kernels

// f32 -> f16 dense copy (features)
__global__ void k_cvt_f16(const float* __restrict__ src, _Float16* __restrict__ dst, int n) {
    int i = blockIdx.x * blockDim.x + threadIdx.x;
    if (i < n) dst[i] = (_Float16)src[i];
}

// Swizzle W (Nout x K, row-major f32) into WMMA-B tile layout (f16).
__global__ void k_swizzle(const float* __restrict__ W, _Float16* __restrict__ sw, int Nout, int K) {
    int idx = blockIdx.x * blockDim.x + threadIdx.x;
    if (idx >= Nout * K) return;
    int tile = idx >> 9, within = idx & 511;
    int lane = within >> 4, rest = within & 15;
    int g = rest >> 3, j = rest & 7;
    int NT = Nout >> 4;
    int kt = tile / NT, nt = tile % NT;
    int n = nt * 16 + (lane & 15);
    int k = kt * 32 + g * 16 + ((lane & 16) ? 8 : 0) + j;
    sw[idx] = (_Float16)W[(size_t)n * K + k];
}

// xq[row] = features[row,:] . wq   (one wave per row)
__global__ void k_dotq(const float* __restrict__ F, const float* __restrict__ wq,
                       float* __restrict__ xq) {
    int wave = blockIdx.x * (blockDim.x >> 5) + (threadIdx.x >> 5);
    int lane = threadIdx.x & 31;
    if (wave >= BN) return;
    const float* row = F + (size_t)wave * DD;
    float acc = 0.f;
#pragma unroll
    for (int j = 0; j < DD / 32; ++j) acc = fmaf(row[lane + 32 * j], wq[lane + 32 * j], acc);
#pragma unroll
    for (int off = 16; off; off >>= 1) acc += __shfl_xor(acc, off, 32);
    if (lane == 0) xq[wave] = acc;
}

// C = A(Mrows x 512 f16) @ Bsw + bias.
// Block: 8 waves, one shared A row-tile staged to LDS (async DMA), 2 col-tiles per wave.
__global__ void __launch_bounds__(256)
k_gemm_bias(const _Float16* __restrict__ A, const _Float16* __restrict__ Bsw,
            const float* __restrict__ bias, float* __restrict__ C,
            int Mrows, int ncols) {
    __shared__ _Float16 sA[16 * DD];                   // 16 KB shared A tile
    const int w    = threadIdx.x >> 5;
    const int lane = threadIdx.x & 31;
    const int NT   = ncols >> 4;
    const int cgs  = NT >> 4;                          // col-groups of 16 tiles
    const int rt   = blockIdx.x / cgs;
    const int cg   = blockIdx.x % cgs;
    const int ct0  = cg * 16 + w * 2;

    // stage A tile: 16x512 f16 = 1024 x 16B chunks, 256 threads x 4
    const _Float16* gsrc = A + (size_t)rt * 16 * DD;
#if USE_ASYNC_LDS
#pragma unroll
    for (int c = 0; c < 4; ++c) {
        int chunk = threadIdx.x + 256 * c;
        __builtin_amdgcn_global_load_async_to_lds_b128(
            as_g4(gsrc + chunk * 8), as_l4(sA + chunk * 8), 0, 0);
    }
    __builtin_amdgcn_s_wait_asynccnt(0);
#else
#pragma unroll
    for (int c = 0; c < 4; ++c) {
        int chunk = threadIdx.x + 256 * c;
        *(v8h*)(sA + chunk * 8) = *(const v8h*)(gsrc + chunk * 8);
    }
#endif
    __syncthreads();

    v8f acc0 = {}, acc1 = {};
#pragma unroll 4
    for (int kt = 0; kt < DD / 32; ++kt) {
        __builtin_prefetch(Bsw + (((size_t)((kt + 1) * NT + ct0)) * 32 + lane) * 16, 0, 3);
        v16h a  = afrag(sA, DD, 0, kt * 32, lane);     // ds_load, shared across both tiles
        v16h b0 = bfrag(Bsw, NT, kt, ct0, lane);
        v16h b1 = bfrag(Bsw, NT, kt, ct0 + 1, lane);
        acc0 = wmma32(a, b0, acc0);
        acc1 = wmma32(a, b1, acc1);
    }
    const int mo = (lane & 16) ? 8 : 0;
#pragma unroll
    for (int t = 0; t < 2; ++t) {
        v8f acc = t ? acc1 : acc0;
        int nc = (ct0 + t) * 16 + (lane & 15);
        float bv = bias ? bias[nc] : 0.f;
#pragma unroll
        for (int r = 0; r < 8; ++r)
            C[(size_t)(rt * 16 + r + mo) * ncols + nc] = acc[r] + bv;
    }
}

// H[:,0] = C0 + P0 (GRU cells with zero partners)
__global__ void k_h0(const float* __restrict__ GI, const float* __restrict__ GHp,
                     const float* __restrict__ bh_c, const float* __restrict__ bi_p,
                     const float* __restrict__ F, float* __restrict__ H) {
    int i = blockIdx.x * blockDim.x + threadIdx.x;
    if (i >= BB * DD) return;
    int b = i / DD, d = i % DD;
    size_t g = (size_t)(b * NN) * TD;
    float r = sigm(GI[g + d] + bh_c[d]);
    float z = sigm(GI[g + DD + d] + bh_c[DD + d]);
    float nn = tanhf(GI[g + 2 * DD + d] + r * bh_c[2 * DD + d]);
    float C0 = (1.f - z) * nn;
    float r2 = sigm(bi_p[d] + GHp[g + d]);
    float z2 = sigm(bi_p[DD + d] + GHp[g + DD + d]);
    float n2 = tanhf(bi_p[2 * DD + d] + r2 * GHp[g + 2 * DD + d]);
    float x0 = F[(size_t)(b * NN) * DD + d];
    float P0 = (1.f - z2) * n2 + z2 * x0;
    H[(size_t)(b * NN) * DD + d] = C0 + P0;
}

// Persistent sequential scan: 4 blocks x 512 threads, block g owns batches [16g,16g+16)
__global__ void __launch_bounds__(512, 1)
k_scan(float* __restrict__ H, const float* __restrict__ F,
       const float* __restrict__ GI, const float* __restrict__ GHp,
       const _Float16* __restrict__ WhCsw, const _Float16* __restrict__ WiPsw,
       const _Float16* __restrict__ Wr0sw, const _Float16* __restrict__ Wr1sw,
       const float* __restrict__ bh_c, const float* __restrict__ bi_p,
       const float* __restrict__ xq, const float* __restrict__ lin_w,
       const float* __restrict__ lin_b, const int* __restrict__ adj,
       const int* __restrict__ smk, const int* __restrict__ omk,
       float* __restrict__ gstage) {
    __shared__ _Float16 s_hs[16 * DD];
    __shared__ _Float16 s_ho[16 * DD];
    __shared__ _Float16 s_Mh[16 * DD];
    __shared__ float    s_Mf[16 * DD];
    __shared__ float    s_wS[16 * NN];
    __shared__ float    s_wO[16 * NN];
    __shared__ float    s_hk[16 * NN];

    const int blk  = blockIdx.x;
    const int w    = threadIdx.x >> 5;     // wave id 0..15
    const int lane = threadIdx.x & 31;
    const int bb   = w;                    // local batch row
    const int b    = blk * 16 + bb;        // global batch
    const float* wk = lin_w + DD;
    const float lbv = lin_b[0];
    float* gc = gstage + (size_t)blk * 2 * 16 * TD;
    float* gp = gc + 16 * TD;

    // prologue: hk[bb][0] = H[b,0,:] . wk
    {
        const float* row = H + (size_t)(b * NN) * DD;
        float acc = 0.f;
#pragma unroll
        for (int j = 0; j < DD / 32; ++j) acc = fmaf(row[lane + 32 * j], wk[lane + 32 * j], acc);
#pragma unroll
        for (int off = 16; off; off >>= 1) acc += __shfl_xor(acc, off, 32);
        if (lane == 0) s_hk[bb * NN + 0] = acc;
    }
    __syncthreads();

    for (int i = 1; i < NN; ++i) {
        // ---- Phase A: masked softmax over history (wave w handles batch bb) ----
        {
            const float xqv = xq[b * NN + i];
            const int* arow = adj + (size_t)(b * NN + i) * NN;
            const int* srow = smk + (size_t)(b * NN + i) * NN;
            const int* orow = omk + (size_t)(b * NN + i) * NN;
            float a[4], m = NEGINF;
#pragma unroll
            for (int c = 0; c < 4; ++c) {
                int n = c * 32 + lane;
                bool valid = (n < i) && (arow[n] != 0);
                a[c] = valid ? (xqv + s_hk[bb * NN + n] + lbv) : NEGINF;
                m = fmaxf(m, a[c]);
            }
#pragma unroll
            for (int off = 16; off; off >>= 1) m = fmaxf(m, __shfl_xor(m, off, 32));
            float e[4], sum = 0.f;
#pragma unroll
            for (int c = 0; c < 4; ++c) { e[c] = (a[c] > 0.5f * NEGINF) ? __expf(a[c] - m) : 0.f; sum += e[c]; }
#pragma unroll
            for (int off = 16; off; off >>= 1) sum += __shfl_xor(sum, off, 32);
            float inv = 1.0f / sum;
#pragma unroll
            for (int c = 0; c < 4; ++c) {
                int n = c * 32 + lane;
                float wv = e[c] * inv;
                s_wS[bb * NN + n] = wv * (float)srow[n];
                s_wO[bb * NN + n] = wv * (float)orow[n];
            }
        }
        __syncwarp();

        // ---- Phase B: hs = sum_n wS*H[b,n,:]; ho = sum_n wO*H[b,n,:] (float4) ----
        {
            float4 accS[4], accO[4];
#pragma unroll
            for (int j = 0; j < 4; ++j) { accS[j] = make_float4(0, 0, 0, 0); accO[j] = make_float4(0, 0, 0, 0); }
            for (int n = 0; n < i; ++n) {
                float ws_v = s_wS[bb * NN + n];
                float wo_v = s_wO[bb * NN + n];
                const float4* Hrow = (const float4*)(H + (size_t)(b * NN + n) * DD);
#pragma unroll
                for (int j = 0; j < 4; ++j) {
                    float4 hv = Hrow[lane + 32 * j];
                    accS[j].x = fmaf(ws_v, hv.x, accS[j].x); accO[j].x = fmaf(wo_v, hv.x, accO[j].x);
                    accS[j].y = fmaf(ws_v, hv.y, accS[j].y); accO[j].y = fmaf(wo_v, hv.y, accO[j].y);
                    accS[j].z = fmaf(ws_v, hv.z, accS[j].z); accO[j].z = fmaf(wo_v, hv.z, accO[j].z);
                    accS[j].w = fmaf(ws_v, hv.w, accS[j].w); accO[j].w = fmaf(wo_v, hv.w, accO[j].w);
                }
            }
#pragma unroll
            for (int j = 0; j < 4; ++j) {
                int d4 = (lane + 32 * j) * 4;
                const float* fs = (const float*)&accS[j];
                const float* fo = (const float*)&accO[j];
#pragma unroll
                for (int c = 0; c < 4; ++c) {
                    s_hs[bb * DD + d4 + c] = (_Float16)fs[c];
                    s_ho[bb * DD + d4 + c] = (_Float16)fo[c];
                }
            }
        }
        __syncthreads();

        // ---- Phase C: M = hs @ Wr0^T + ho @ Wr1^T (16x512, WMMA, shared A-frag) ----
        {
            const int ct0 = w * 2;                        // 32 col-tiles, 2 per wave
            v8f acc0 = {}, acc1 = {};
#pragma unroll 4
            for (int kt = 0; kt < DD / 32; ++kt) {
                v16h a  = afrag(s_hs, DD, 0, kt * 32, lane);
                v16h b0 = bfrag(Wr0sw, DD / 16, kt, ct0, lane);
                v16h b1 = bfrag(Wr0sw, DD / 16, kt, ct0 + 1, lane);
                acc0 = wmma32(a, b0, acc0);
                acc1 = wmma32(a, b1, acc1);
            }
#pragma unroll 4
            for (int kt = 0; kt < DD / 32; ++kt) {
                v16h a  = afrag(s_ho, DD, 0, kt * 32, lane);
                v16h b0 = bfrag(Wr1sw, DD / 16, kt, ct0, lane);
                v16h b1 = bfrag(Wr1sw, DD / 16, kt, ct0 + 1, lane);
                acc0 = wmma32(a, b0, acc0);
                acc1 = wmma32(a, b1, acc1);
            }
            const int mo = (lane & 16) ? 8 : 0;
#pragma unroll
            for (int t = 0; t < 2; ++t) {
                v8f acc = t ? acc1 : acc0;
                int nc = (ct0 + t) * 16 + (lane & 15);
#pragma unroll
                for (int r = 0; r < 8; ++r) {
                    s_Mf[(r + mo) * DD + nc] = acc[r];
                    s_Mh[(r + mo) * DD + nc] = (_Float16)acc[r];
                }
            }
        }
        __syncthreads();

        // ---- Phase D: gh_c = M @ Wh_c^T ; gi_p = M @ Wi_p^T (16x1536 each, WMMA) ----
        // One A-fragment feeds 4 WMMAs (2 matrices x 2 col-tiles).
        {
#pragma unroll
            for (int tp = 0; tp < 3; ++tp) {
                int ct0 = w + 32 * tp;
                int ct1 = ct0 + 16;
                v8f ac0 = {}, ac1 = {}, ap0 = {}, ap1 = {};
#pragma unroll 4
                for (int kt = 0; kt < DD / 32; ++kt) {
                    __builtin_prefetch(WhCsw + (((size_t)((kt + 1) * (TD / 16) + ct0)) * 32 + lane) * 16, 0, 3);
                    v16h a   = afrag(s_Mh, DD, 0, kt * 32, lane);
                    v16h bc0 = bfrag(WhCsw, TD / 16, kt, ct0, lane);
                    v16h bc1 = bfrag(WhCsw, TD / 16, kt, ct1, lane);
                    v16h bp0 = bfrag(WiPsw, TD / 16, kt, ct0, lane);
                    v16h bp1 = bfrag(WiPsw, TD / 16, kt, ct1, lane);
                    ac0 = wmma32(a, bc0, ac0);
                    ac1 = wmma32(a, bc1, ac1);
                    ap0 = wmma32(a, bp0, ap0);
                    ap1 = wmma32(a, bp1, ap1);
                }
                const int mo = (lane & 16) ? 8 : 0;
                int nc0 = ct0 * 16 + (lane & 15);
                int nc1 = ct1 * 16 + (lane & 15);
#pragma unroll
                for (int r = 0; r < 8; ++r) {
                    gc[(size_t)(r + mo) * TD + nc0] = ac0[r];
                    gc[(size_t)(r + mo) * TD + nc1] = ac1[r];
                    gp[(size_t)(r + mo) * TD + nc0] = ap0[r];
                    gp[(size_t)(r + mo) * TD + nc1] = ap1[r];
                }
            }
        }
        __threadfence();
        __syncthreads();

        // ---- Phase E: GRU pointwise (float4), H[:,i] = C + P, hk update ----
        {
            float hkacc = 0.f;
            size_t gi = (size_t)(b * NN + i) * TD;
            size_t gx = (size_t)(b * NN + i) * DD;
#pragma unroll
            for (int j = 0; j < 4; ++j) {
                int d4 = (lane + 32 * j) * 4;
                float4 ir4  = *(const float4*)(GI + gi + d4);
                float4 iz4  = *(const float4*)(GI + gi + DD + d4);
                float4 in4  = *(const float4*)(GI + gi + 2 * DD + d4);
                float4 gcr  = *(const float4*)(gc + (size_t)bb * TD + d4);
                float4 gcz  = *(const float4*)(gc + (size_t)bb * TD + DD + d4);
                float4 gcn  = *(const float4*)(gc + (size_t)bb * TD + 2 * DD + d4);
                float4 bhr  = *(const float4*)(bh_c + d4);
                float4 bhz  = *(const float4*)(bh_c + DD + d4);
                float4 bhn  = *(const float4*)(bh_c + 2 * DD + d4);
                float4 gpr  = *(const float4*)(gp + (size_t)bb * TD + d4);
                float4 gpz  = *(const float4*)(gp + (size_t)bb * TD + DD + d4);
                float4 gpn  = *(const float4*)(gp + (size_t)bb * TD + 2 * DD + d4);
                float4 bir  = *(const float4*)(bi_p + d4);
                float4 biz  = *(const float4*)(bi_p + DD + d4);
                float4 bin  = *(const float4*)(bi_p + 2 * DD + d4);
                float4 ghr  = *(const float4*)(GHp + gi + d4);
                float4 ghz  = *(const float4*)(GHp + gi + DD + d4);
                float4 ghn  = *(const float4*)(GHp + gi + 2 * DD + d4);
                float4 xv4  = *(const float4*)(F + gx + d4);
                float4 wk4  = *(const float4*)(wk + d4);
                float4 hout;
#pragma unroll
                for (int c = 0; c < 4; ++c) {
                    float ir = ((const float*)&ir4)[c], iz = ((const float*)&iz4)[c], inn = ((const float*)&in4)[c];
                    float hr = ((const float*)&gcr)[c] + ((const float*)&bhr)[c];
                    float hz = ((const float*)&gcz)[c] + ((const float*)&bhz)[c];
                    float hn = ((const float*)&gcn)[c] + ((const float*)&bhn)[c];
                    float r  = sigm(ir + hr);
                    float z  = sigm(iz + hz);
                    float nn = tanhf(inn + r * hn);
                    float Mv = s_Mf[bb * DD + d4 + c];
                    float Cc = (1.f - z) * nn + z * Mv;
                    float pr = ((const float*)&gpr)[c] + ((const float*)&bir)[c] + ((const float*)&ghr)[c];
                    float pz = ((const float*)&gpz)[c] + ((const float*)&biz)[c] + ((const float*)&ghz)[c];
                    float pnl = ((const float*)&gpn)[c] + ((const float*)&bin)[c];
                    float r2 = sigm(pr);
                    float z2 = sigm(pz);
                    float n2 = tanhf(pnl + r2 * ((const float*)&ghn)[c]);
                    float xv = ((const float*)&xv4)[c];
                    float Pp = (1.f - z2) * n2 + z2 * xv;
                    float h  = Cc + Pp;
                    ((float*)&hout)[c] = h;
                    hkacc = fmaf(h, ((const float*)&wk4)[c], hkacc);
                }
                *(float4*)(H + gx + d4) = hout;
            }
#pragma unroll
            for (int off = 16; off; off >>= 1) hkacc += __shfl_xor(hkacc, off, 32);
            if (lane == 0) s_hk[bb * NN + i] = hkacc;
        }
        __syncthreads();
    }
}

// ---------------------------------------------------------------- launch
extern "C" void kernel_launch(void* const* d_in, const int* in_sizes, int n_in,
                              void* d_out, int out_size, void* d_ws, size_t ws_size,
                              hipStream_t stream) {
    const float* F     = (const float*)d_in[0];
    const float* Wi_c  = (const float*)d_in[1];
    const float* Wh_c  = (const float*)d_in[2];
    const float* bi_c  = (const float*)d_in[3];
    const float* bh_c  = (const float*)d_in[4];
    const float* Wi_p  = (const float*)d_in[5];
    const float* Wh_p  = (const float*)d_in[6];
    const float* bi_p  = (const float*)d_in[7];
    const float* bh_p  = (const float*)d_in[8];
    const float* lin_w = (const float*)d_in[9];
    const float* lin_b = (const float*)d_in[10];
    const float* Wr0   = (const float*)d_in[11];
    const float* Wr1   = (const float*)d_in[12];
    const int*   adj   = (const int*)d_in[13];
    const int*   smk   = (const int*)d_in[14];
    const int*   omk   = (const int*)d_in[15];
    float* H = (float*)d_out;

    size_t off = 0;
    char* base = (char*)d_ws;
    auto carve = [&](size_t bytes) -> void* {
        off = (off + 255) & ~(size_t)255;
        void* p = base + off;
        off += bytes;
        return p;
    };
    _Float16* feat16 = (_Float16*)carve((size_t)BN * DD * 2);
    _Float16* WiCsw  = (_Float16*)carve((size_t)TD * DD * 2);
    _Float16* WhPsw  = (_Float16*)carve((size_t)TD * DD * 2);
    _Float16* WhCsw  = (_Float16*)carve((size_t)TD * DD * 2);
    _Float16* WiPsw  = (_Float16*)carve((size_t)TD * DD * 2);
    _Float16* Wr0sw  = (_Float16*)carve((size_t)DD * DD * 2);
    _Float16* Wr1sw  = (_Float16*)carve((size_t)DD * DD * 2);
    float*    GI     = (float*)carve((size_t)BN * TD * 4);
    float*    GHp    = (float*)carve((size_t)BN * TD * 4);
    float*    xq     = (float*)carve((size_t)BN * 4);
    float*    gstage = (float*)carve((size_t)4 * 2 * 16 * TD * 4);
    (void)ws_size; (void)n_in; (void)in_sizes; (void)out_size;

    // 1) conversions + swizzles
    k_cvt_f16<<<(BN * DD + 255) / 256, 256, 0, stream>>>(F, feat16, BN * DD);
    k_swizzle<<<(TD * DD + 255) / 256, 256, 0, stream>>>(Wi_c, WiCsw, TD, DD);
    k_swizzle<<<(TD * DD + 255) / 256, 256, 0, stream>>>(Wh_p, WhPsw, TD, DD);
    k_swizzle<<<(TD * DD + 255) / 256, 256, 0, stream>>>(Wh_c, WhCsw, TD, DD);
    k_swizzle<<<(TD * DD + 255) / 256, 256, 0, stream>>>(Wi_p, WiPsw, TD, DD);
    k_swizzle<<<(DD * DD + 255) / 256, 256, 0, stream>>>(Wr0, Wr0sw, DD, DD);
    k_swizzle<<<(DD * DD + 255) / 256, 256, 0, stream>>>(Wr1, Wr1sw, DD, DD);

    // 2) xq = features . wq
    k_dotq<<<BN / 8, 256, 0, stream>>>(F, lin_w, xq);

    // 3) whole-chip precompute GEMMs (WMMA, async LDS A-staging)
    {
        int blocks = (BN / 16) * ((TD / 16) / 16);     // 512 row-tiles x 6 col-groups
        k_gemm_bias<<<blocks, 256, 0, stream>>>(feat16, WiCsw, bi_c, GI, BN, TD);
        k_gemm_bias<<<blocks, 256, 0, stream>>>(feat16, WhPsw, bh_p, GHp, BN, TD);
    }

    // 4) H[:,0]
    k_h0<<<(BB * DD + 255) / 256, 256, 0, stream>>>(GI, GHp, bh_c, bi_p, F, H);

    // 5) persistent sequential scan: 4 batch-groups of 16
    k_scan<<<4, 512, 0, stream>>>(H, F, GI, GHp, WhCsw, WiPsw, Wr0sw, Wr1sw,
                                  bh_c, bi_p, xq, lin_w, lin_b, adj, smk, omk, gstage);
}
